// TemplateEmbedder_43946105373396
// MI455X (gfx1250) — compile-verified, matrix operand order checked by reading, not run on hardware
//
#include <hip/hip_runtime.h>
#include <hip/hip_bf16.h>
#include <math.h>

typedef __bf16 bf16;
typedef bf16  v16bf __attribute__((ext_vector_type(16)));
typedef float v8f   __attribute__((ext_vector_type(8)));

#define DEV static __device__ __forceinline__

// k-index inside a 16-bit WMMA A/B fragment for lane `lane`, element e (0..15).
DEV int frag_k(int lane, int e) {
  int v = e >> 1;
  return ((v >> 2) << 4) | ((lane >> 4) << 3) | ((v & 3) << 1) | (e & 1);
}
DEV float sigmoidf(float x) { return 1.0f / (1.0f + expf(-x)); }

// ---------------- generic GEMM: C = act(alpha * A[M,K] @ B[K,N] + bias) -------------
// fp32 in/out, bf16 WMMA with f32 accumulate. act: 0=none 1=relu 2=sigmoid
// Main K loop is guard-free; K tail handled with in-bounds clamped loads + select.
// N edge handled by clamping the load column and guarding only the store.
__global__ void gemm_kernel(const float* __restrict__ A, const float* __restrict__ B,
                            const float* __restrict__ bias, float* __restrict__ C,
                            int M, int N, int K, float alpha, int act) {
  int lane = threadIdx.x & 31;
  int wave = threadIdx.x >> 5;
  int mt = blockIdx.x * 4 + wave;
  if (mt * 16 >= M) return;
  int row0 = mt * 16;
  int arow = row0 + (lane & 15);
  int coln = blockIdx.y * 16 + (lane & 15);
  int colL = (coln < N) ? coln : (N - 1);       // clamped: loads always in bounds
  const float* __restrict__ Ar = A + (long)arow * K;
  v8f acc = {0.f, 0.f, 0.f, 0.f, 0.f, 0.f, 0.f, 0.f};
  int kmain = K & ~31;
  for (int kk = 0; kk < kmain; kk += 32) {
    __builtin_prefetch(Ar + kk + 64, 0, 0);     // global_prefetch, speculative
    v16bf a, b;
#pragma unroll
    for (int e = 0; e < 16; ++e) {
      int k = kk + frag_k(lane, e);
      a[e] = (bf16)Ar[k];
      b[e] = (bf16)B[(long)k * N + colL];
    }
    acc = __builtin_amdgcn_wmma_f32_16x16x32_bf16(false, a, false, b, (short)0, acc,
                                                  false, false);
  }
  if (kmain < K) {                               // K tail (K=57, 88): selects, no branches
    v16bf a, b;
#pragma unroll
    for (int e = 0; e < 16; ++e) {
      int k = kmain + frag_k(lane, e);
      int kc = (k < K) ? k : (K - 1);
      float av = Ar[kc];
      float bv = B[(long)kc * N + colL];
      a[e] = (bf16)((k < K) ? av : 0.f);
      b[e] = (bf16)((k < K) ? bv : 0.f);
    }
    acc = __builtin_amdgcn_wmma_f32_16x16x32_bf16(false, a, false, b, (short)0, acc,
                                                  false, false);
  }
  if (coln < N) {
    float bv = bias ? bias[coln] : 0.f;
#pragma unroll
    for (int r = 0; r < 8; ++r) {
      int row = row0 + r + ((lane >> 4) << 3);
      float v = alpha * acc[r] + bv;
      if (act == 1) v = fmaxf(v, 0.f);
      else if (act == 2) v = sigmoidf(v);
      C[(long)row * N + coln] = v;
    }
  }
}

// ---------------- triangle attention core --------------------------------------------
// One block per (fixed index f, head h, template s). 128 threads (4 waves).
// starting: fixed i, rows=j, cols=k, bias tb[j,k,h].  ending: fixed j, rows=i,
// cols=k, bias tb[k,i,h]. Q pre-scaled by DH^-0.5, G pre-sigmoided.
// q/k staged in LDS as zero-padded bf16 [128][32] so fragment builds are guard-free.
__global__ void tri_attn_kernel(const float* __restrict__ Q, const float* __restrict__ Kt,
                                const float* __restrict__ V, const float* __restrict__ G,
                                const float* __restrict__ TB, float* __restrict__ O,
                                int starting) {
  __shared__ bf16 qs[128][32];
  __shared__ bf16 ks[128][32];
  __shared__ bf16 vs[128][16];
  __shared__ float lg[64][128];   // logits / probs for a 64-row half
  int f = blockIdx.x, h = blockIdx.y, s = blockIdx.z;
  int tid = threadIdx.x;
  int lane = tid & 31, wave = tid >> 5;
  long base = (long)s * 1048576 + (starting ? f * 8192 : f * 64) + h * 16;
  int rowStride = starting ? 64 : 8192;
  long tbBase = (long)s * 65536;

  for (int idx = tid; idx < 2048; idx += 128) {
    int r = idx >> 4, d = idx & 15;
    long g = base + (long)r * rowStride + d;
    qs[r][d] = (bf16)Q[g];
    qs[r][d + 16] = (bf16)0.f;
    ks[r][d] = (bf16)Kt[g];
    ks[r][d + 16] = (bf16)0.f;
    vs[r][d] = (bf16)V[g];
  }
  __syncthreads();

  for (int half = 0; half < 2; ++half) {
    // logits = Q K^T + bias for rows [half*64, half*64+64)
    for (int t = 0; t < 8; ++t) {
      int ct = wave * 2 + (t & 1);
      int rt = t >> 1;
      int grow0 = half * 64 + rt * 16;
      int ar = grow0 + (lane & 15);
      int bc = ct * 16 + (lane & 15);
      v16bf a, b;
#pragma unroll
      for (int e = 0; e < 16; ++e) {
        int k = frag_k(lane, e);           // contraction dim d, zero-padded 16->32
        a[e] = qs[ar][k];
        b[e] = ks[bc][k];
      }
      v8f acc = {0.f, 0.f, 0.f, 0.f, 0.f, 0.f, 0.f, 0.f};
      acc = __builtin_amdgcn_wmma_f32_16x16x32_bf16(false, a, false, b, (short)0, acc,
                                                    false, false);
      int col = ct * 16 + (lane & 15);
#pragma unroll
      for (int r = 0; r < 8; ++r) {
        int grow = grow0 + r + ((lane >> 4) << 3);
        long bi = tbBase + (long)(starting ? (grow * 128 + col) : (col * 128 + grow)) * 4 + h;
        lg[grow - half * 64][col] = acc[r] + TB[bi];
      }
    }
    __syncthreads();
    if (tid < 64) {
      float m = -1e30f;
      for (int c = 0; c < 128; ++c) m = fmaxf(m, lg[tid][c]);
      float ssum = 0.f;
      for (int c = 0; c < 128; ++c) { float e = expf(lg[tid][c] - m); lg[tid][c] = e; ssum += e; }
      float inv = 1.f / ssum;
      for (int c = 0; c < 128; ++c) lg[tid][c] *= inv;
    }
    __syncthreads();
    // O = P @ V (rows of this half), gated write-out
    {
      int rt = wave;                     // 4 row tiles, 4 waves
      int lr = rt * 16 + (lane & 15);    // local row in lg
      int col = lane & 15;               // d
      v8f acc = {0.f, 0.f, 0.f, 0.f, 0.f, 0.f, 0.f, 0.f};
      for (int kk = 0; kk < 128; kk += 32) {
        v16bf a, b;
#pragma unroll
        for (int e = 0; e < 16; ++e) {
          int k = kk + frag_k(lane, e);
          a[e] = (bf16)lg[lr][k];
          b[e] = vs[k][col];
        }
        acc = __builtin_amdgcn_wmma_f32_16x16x32_bf16(false, a, false, b, (short)0, acc,
                                                      false, false);
      }
#pragma unroll
      for (int r = 0; r < 8; ++r) {
        int grow = half * 64 + rt * 16 + r + ((lane >> 4) << 3);
        long gi = base + (long)grow * rowStride + col;
        O[gi] = acc[r] * G[gi];
      }
    }
    __syncthreads();
  }
}

// ---------------- triangle multiplication: per-channel 128x128x128 ------------------
// outgoing: x[i,j,c] = sum_k a[i,k,c] b[j,k,c]; incoming: sum_k a[k,i,c] b[k,j,c]
__global__ void tri_mult_kernel(const float* __restrict__ A, const float* __restrict__ B,
                                float* __restrict__ X, int outgoing) {
  int lane = threadIdx.x & 31;
  int tile = blockIdx.x;            // 0..63 -> 8x8 tiles
  int it = tile >> 3, jt = tile & 7;
  int c = blockIdx.y, s = blockIdx.z;
  long cbase = (long)s * 1048576 + c;
  long rS = outgoing ? 8192 : 64;   // stride of the free index
  long kS = outgoing ? 64 : 8192;   // stride of the contraction index
  int ar = it * 16 + (lane & 15);
  int bc = jt * 16 + (lane & 15);
  v8f acc = {0.f, 0.f, 0.f, 0.f, 0.f, 0.f, 0.f, 0.f};
  for (int kk = 0; kk < 128; kk += 32) {
    v16bf a, b;
#pragma unroll
    for (int e = 0; e < 16; ++e) {
      int k = kk + frag_k(lane, e);
      a[e] = (bf16)A[cbase + (long)ar * rS + (long)k * kS];
      b[e] = (bf16)B[cbase + (long)bc * rS + (long)k * kS];
    }
    acc = __builtin_amdgcn_wmma_f32_16x16x32_bf16(false, a, false, b, (short)0, acc,
                                                  false, false);
  }
#pragma unroll
  for (int r = 0; r < 8; ++r) {
    int row = it * 16 + r + ((lane >> 4) << 3);
    X[(long)s * 1048576 + (long)(row * 128 + bc) * 64 + c] = acc[r];
  }
}

// ---------------- LayerNorm over last dim (64), one thread per row ------------------
__global__ void ln_kernel(const float* __restrict__ x, const float* __restrict__ g,
                          const float* __restrict__ b, float* __restrict__ y, int rows) {
  int r = blockIdx.x * blockDim.x + threadIdx.x;
  if (r >= rows) return;
  const float* xr = x + (long)r * 64;
  float m = 0.f;
  for (int c = 0; c < 64; ++c) m += xr[c];
  m *= (1.f / 64.f);
  float v = 0.f;
  for (int c = 0; c < 64; ++c) { float d = xr[c] - m; v += d * d; }
  v *= (1.f / 64.f);
  float inv = rsqrtf(v + 1e-5f);
  float* yr = y + (long)r * 64;
  for (int c = 0; c < 64; ++c) yr[c] = (xr[c] - m) * inv * g[c] + b[c];
}

// ---------------- elementwise helpers ------------------------------------------------
__global__ void mul_kernel(float* __restrict__ o, const float* __restrict__ a,
                           const float* __restrict__ b, int n) {
  int i = blockIdx.x * 256 + threadIdx.x;
  if (i < n) o[i] = a[i] * b[i];
}
__global__ void add_kernel(float* __restrict__ z, const float* __restrict__ x, int n) {
  int i = blockIdx.x * 256 + threadIdx.x;
  if (i < n) z[i] += x[i];
}
__global__ void fmaadd_kernel(float* __restrict__ z, const float* __restrict__ g,
                              const float* __restrict__ x, int n) {
  int i = blockIdx.x * 256 + threadIdx.x;
  if (i < n) z[i] += g[i] * x[i];
}

// ---------------- fused pointwise attention over templates --------------------------
// One block per (i,j) position; 256 threads = 256 flat (c,h) channels.
__global__ void pointwise_kernel(const float* __restrict__ pair, const float* __restrict__ temp,
                                 const float* __restrict__ wq, const float* __restrict__ wkv,
                                 const float* __restrict__ wo, const float* __restrict__ bo,
                                 float* __restrict__ out) {
  __shared__ float pr[128];
  __shared__ float tp[4][64];
  __shared__ float ov[256];
  int pos = blockIdx.x;           // 0..16383
  int t = threadIdx.x;            // 0..255
  if (t < 128) pr[t] = pair[(long)pos * 128 + t];
  { int s = t >> 6, c = t & 63; tp[s][c] = temp[(long)s * 1048576 + (long)pos * 64 + c]; }
  __syncthreads();
  float q = 0.f;
  for (int c = 0; c < 128; ++c) q += pr[c] * wq[c * 256 + t];
  float kv[4][2];
  for (int s = 0; s < 4; ++s) {
    float kk = 0.f, vv = 0.f;
    for (int c = 0; c < 64; ++c) {
      float tv = tp[s][c];
      kk += tv * wkv[c * 512 + t];
      vv += tv * wkv[c * 512 + 256 + t];
    }
    kv[s][0] = kk; kv[s][1] = vv;
  }
  float l[4], m = -1e30f;
  for (int s = 0; s < 4; ++s) { l[s] = 0.125f * q * kv[s][0]; m = fmaxf(m, l[s]); }
  float ssum = 0.f, o = 0.f;
  for (int s = 0; s < 4; ++s) { float e = expf(l[s] - m); ssum += e; o += e * kv[s][1]; }
  ov[t] = o / ssum;
  __syncthreads();
  if (t < 128) {
    float acc = bo[t] + pr[t];
    for (int u = 0; u < 256; ++u) acc += ov[u] * wo[u * 128 + t];
    out[(long)pos * 128 + t] = acc;
  }
}

// ---------------- host orchestration -------------------------------------------------
extern "C" void kernel_launch(void* const* d_in, const int* in_sizes, int n_in,
                              void* d_out, int out_size, void* d_ws, size_t ws_size,
                              hipStream_t stream) {
  (void)in_sizes; (void)n_in; (void)out_size; (void)ws_size;
  auto F = [&](int i) { return (const float*)d_in[i]; };
  float* ws = (float*)d_ws;
  const long B4M = 4194304;   // one [4,128,128,64] f32 tensor
  float* T0  = ws;
  float* Z   = ws + 1 * B4M;
  float* ZLN = ws + 2 * B4M;
  float* Qb  = ws + 3 * B4M;
  float* Kb  = ws + 4 * B4M;
  float* Vb  = ws + 5 * B4M;
  float* Gb  = ws + 6 * B4M;
  float* Xb  = ws + 7 * B4M;
  float* TBb = ws + 8 * B4M;            // 262144 floats
  float* ANG = ws + 8 * B4M + 262144;   // 131072 floats
  float* Hb  = Qb;                      // transition hidden, spans Qb..Kb (8M floats)

  const int NT = 65536;                 // S*128*128 rows
  const int n4 = 4194304;

  auto gemm = [&](const float* A, const float* Bm, const float* bias, float* C,
                  int M, int N, int K, float alpha, int act) {
    dim3 g((M + 63) / 64, (N + 15) / 16);
    gemm_kernel<<<g, 128, 0, stream>>>(A, Bm, bias, C, M, N, K, alpha, act);
  };
  auto ln = [&](const float* x, int gi, int bi, float* y) {
    ln_kernel<<<256, 256, 0, stream>>>(x, F(gi), F(bi), y, NT);
  };
  auto eadd = [&](float* z, const float* x) { add_kernel<<<16384, 256, 0, stream>>>(z, x, n4); };
  auto emul = [&](float* o, const float* a, const float* b) { mul_kernel<<<16384, 256, 0, stream>>>(o, a, b, n4); };
  auto efma = [&](float* z, const float* g, const float* x) { fmaadd_kernel<<<16384, 256, 0, stream>>>(z, g, x, n4); };

  // msa_out[:256] = msa_rep (8,388,608 floats)
  hipMemcpyAsync(d_out, d_in[0], (size_t)8388608 * 4, hipMemcpyDeviceToDevice, stream);
  // angle MLP -> msa_out[256:260]
  gemm(F(2), F(4), F(5), ANG, 512, 256, 57, 1.f, 1);
  gemm(ANG, F(6), F(7), (float*)d_out + 8388608, 512, 256, 256, 1.f, 0);
  // temp = temp_pair_feats @ pairf_w + pairf_b
  gemm(F(3), F(8), F(9), T0, NT, 64, 88, 1.f, 0);
  hipMemcpyAsync(Z, T0, (size_t)n4 * 4, hipMemcpyDeviceToDevice, stream);

  for (int blk = 0; blk < 2; ++blk) {
    int pb = 10 + blk * 58;
    // triangle attention: starting (param offset 0) then ending (offset 10)
    for (int at = 0; at < 2; ++at) {
      int P = pb + at * 10;   // ln_g,ln_b,wq,wk,wv,wb,wg,bg,wo,bo
      int starting = (at == 0);
      ln(Z, P + 0, P + 1, ZLN);
      gemm(ZLN, F(P + 2), nullptr, Qb, NT, 64, 64, 0.25f, 0);   // q * DH^-0.5
      gemm(ZLN, F(P + 3), nullptr, Kb, NT, 64, 64, 1.f, 0);     // k
      gemm(ZLN, F(P + 4), nullptr, Vb, NT, 64, 64, 1.f, 0);     // v
      gemm(ZLN, F(P + 5), nullptr, TBb, NT, 4, 64, 1.f, 0);     // triangle bias
      gemm(ZLN, F(P + 6), F(P + 7), Gb, NT, 64, 64, 1.f, 2);    // sigmoid gate
      tri_attn_kernel<<<dim3(128, 4, 4), 128, 0, stream>>>(Qb, Kb, Vb, Gb, TBb, Xb, starting);
      gemm(Xb, F(P + 8), F(P + 9), Vb, NT, 64, 64, 1.f, 0);     // out proj
      eadd(Z, Vb);
    }
    // triangle multiplication: outgoing (offset 20) then incoming (offset 36)
    for (int mt = 0; mt < 2; ++mt) {
      int P = pb + 20 + mt * 16;  // ln_g,ln_b,ln2_g,ln2_b,wa,wga,wb,wgb,wo,wgo,ba,bga,bb,bgb,bo,bgo
      int outgoing = (mt == 0);
      ln(Z, P + 0, P + 1, ZLN);
      gemm(ZLN, F(P + 5), F(P + 11), Qb, NT, 64, 64, 1.f, 2);   // sigmoid(z wga + bga)
      gemm(ZLN, F(P + 4), F(P + 10), Kb, NT, 64, 64, 1.f, 0);   // z wa + ba
      emul(Qb, Qb, Kb);                                          // a
      gemm(ZLN, F(P + 7), F(P + 13), Kb, NT, 64, 64, 1.f, 2);   // sigmoid(z wgb + bgb)
      gemm(ZLN, F(P + 6), F(P + 12), Vb, NT, 64, 64, 1.f, 0);   // z wb + bb
      emul(Kb, Kb, Vb);                                          // b
      tri_mult_kernel<<<dim3(64, 64, 4), 32, 0, stream>>>(Qb, Kb, Xb, outgoing);
      ln(Xb, P + 2, P + 3, Vb);                                  // ln2(x)
      gemm(Vb, F(P + 8), F(P + 14), Xb, NT, 64, 64, 1.f, 0);    // xln wo + bo
      gemm(ZLN, F(P + 9), F(P + 15), Gb, NT, 64, 64, 1.f, 2);   // sigmoid(z wgo + bgo)
      efma(Z, Gb, Xb);
    }
    // transition (offset 52): ln_g,ln_b,w1,b1,w2,b2
    {
      int P = pb + 52;
      ln(Z, P + 0, P + 1, ZLN);
      gemm(ZLN, F(P + 2), F(P + 3), Hb, NT, 128, 64, 1.f, 1);
      gemm(Hb, F(P + 4), F(P + 5), Xb, NT, 64, 128, 1.f, 0);
      eadd(Z, Xb);
    }
  }
  // stack LN + outer residual: temp_final = t0 + LN(z)
  ln(Z, 126, 127, ZLN);
  eadd(T0, ZLN);
  // pointwise template attention -> pair_out
  pointwise_kernel<<<16384, 256, 0, stream>>>(F(1), T0, F(128), F(129), F(130), F(131),
                                              (float*)d_out + 8519680);
}
